// FourierKAN_Predictor_54554674594313
// MI455X (gfx1250) — compile-verified, matrix operand order checked by reading, not run on hardware
//
#include <hip/hip_runtime.h>
#include <hip/hip_bf16.h>
#include <cstdint>

typedef __attribute__((ext_vector_type(16))) __bf16 v16bf;
typedef __attribute__((ext_vector_type(8)))  float  v8f;

#define KAN_BATCH 4096
#define BM 128
#define BN 128
#define A_STRIDE 40   // halves per LDS feature row (80B: 16B aligned, conflict-free reads)

union BFragU { uint4 q[2]; v16bf v; };

// Round-to-nearest-even f32->bf16 on the integer pipe, pair-packed with one
// v_perm_b32 (splice the two rounded high halves) instead of shift/and/or.
static __device__ __forceinline__ unsigned pk_bf16(float lo, float hi) {
    unsigned a = __builtin_bit_cast(unsigned, lo);
    unsigned b = __builtin_bit_cast(unsigned, hi);
    a += 0x7fffu + ((a >> 16) & 1u);
    b += 0x7fffu + ((b >> 16) & 1u);
#if __has_builtin(__builtin_amdgcn_perm)
    // dest = { b.byte3, b.byte2, a.byte3, a.byte2 }  (sel 0-3 -> src1, 4-7 -> src0)
    return __builtin_amdgcn_perm(b, a, 0x07060302u);
#else
    return (a >> 16) | (b & 0xffff0000u);
#endif
}

// ---------------------------------------------------------------------------
// Repack C (2, OUT, IN, 8) f32  ->  bf16 WMMA-B fragments:
//   W[kb][nb][lane][16 halves],  kb = K/32 (K = i*16 + d*8 + g), nb = j/16
// Half->K_local map (ISA 16-bit B layout): K = h + (h>=8?8:0) + (lane>=16?8:0)
// => lane<16 holds d=0 (cos) rows, lane>=16 holds d=1 (sin) rows; each 8-half
//    group is one contiguous 32B read of C[d, j, i, 0..7].
// ---------------------------------------------------------------------------
__global__ __launch_bounds__(256) void convert_weights(
    const float* __restrict__ C, unsigned short* __restrict__ W,
    int IN, int OUT) {
    long tid   = (long)blockIdx.x * blockDim.x + threadIdx.x;
    long total = (long)IN * OUT;             // (16*IN/32) * (OUT/16) * 32 threads
    if (tid >= total) return;

    int  lane    = (int)(tid & 31);
    long rest    = tid >> 5;
    int  nbCount = OUT >> 4;
    int  nb      = (int)(rest % nbCount);
    long kb      = rest / nbCount;

    int  j  = nb * 16 + (lane & 15);
    int  d0 = (lane >= 16) ? 1 : 0;
    long i0 = kb * 2;

    const float* s0 = C + (((long)d0 * OUT + j) * IN + i0) * 8;  // channel i0
    const float* s1 = s0 + 8;                                    // channel i0+1

    float4 a0 = ((const float4*)s0)[0];
    float4 a1 = ((const float4*)s0)[1];
    float4 b0 = ((const float4*)s1)[0];
    float4 b1 = ((const float4*)s1)[1];

    uint4 o0, o1;
    o0.x = pk_bf16(a0.x, a0.y);
    o0.y = pk_bf16(a0.z, a0.w);
    o0.z = pk_bf16(a1.x, a1.y);
    o0.w = pk_bf16(a1.z, a1.w);
    o1.x = pk_bf16(b0.x, b0.y);
    o1.y = pk_bf16(b0.z, b0.w);
    o1.z = pk_bf16(b1.x, b1.y);
    o1.w = pk_bf16(b1.z, b1.w);

    uint4* dst = (uint4*)(W + (((kb * (long)nbCount) + nb) * 32 + lane) * 16);
    dst[0] = o0;
    dst[1] = o1;
}

// ---------------------------------------------------------------------------
// Fused FourierKAN layer: Y = Fourier(X) @ W + bias
//   X: (4096, IN) f32    W: packed bf16 fragments    Y: (4096, N) f32
// Block: 128x128 tile, 8 wave32s (2 M-waves x 4 N-waves), wave = 64x32,
// per wave: 4x2 grid of v_wmma_f32_16x16x32_bf16 accumulators.
// K loop: 2 input channels per step; features built on the fly in LDS
// (double-buffered, one barrier per step); X loads pipelined one full
// iteration ahead so their latency hides behind a K-step of WMMAs.
// ---------------------------------------------------------------------------
__global__ __launch_bounds__(256) void kan_layer_wmma(
    const float* __restrict__ X, const unsigned short* __restrict__ W,
    const float* __restrict__ bias, float* __restrict__ Y,
    int IN, int N) {
    __shared__ unsigned short As[2][BM * A_STRIDE];   // 20.5 KB

    const int tid   = threadIdx.x;
    const int wave  = tid >> 5;
    const int lane  = tid & 31;
    const int mwave = wave >> 2;            // 0..1 -> 64 rows each
    const int nwave = wave & 3;             // 0..3 -> 32 cols each
    const int row0  = blockIdx.y * BM;
    const int col0  = blockIdx.x * BN;
    const int nbCount = N >> 4;

    const int fr = tid & 127;               // feature row within tile
    const int fc = tid >> 7;                // feature channel within pair (0/1)
    const float* xp = X + (long)(row0 + fr) * IN + fc;   // this thread's x column walk

    v8f acc[4][2];
#pragma unroll
    for (int mt = 0; mt < 4; ++mt)
#pragma unroll
        for (int nt = 0; nt < 2; ++nt) acc[mt][nt] = (v8f)0.0f;

    const int nsteps = IN >> 1;

    // Build 8 cos + 8 sin bf16 features from x -> LDS buffer b.
    auto featBody = [&](float x, int b) {
        float s1, c1;
        __sincosf(x, &s1, &c1);
        float cf[8], sf[8];
        cf[0] = c1; sf[0] = s1;
        float ck = c1, sk = s1;
#pragma unroll
        for (int k = 1; k < 8; ++k) {       // angle-addition recurrence
            float cn = ck * c1 - sk * s1;
            float sn = sk * c1 + ck * s1;
            ck = cn; sk = sn;
            cf[k] = ck; sf[k] = sk;
        }
        uint4 q0, q1;
        q0.x = pk_bf16(cf[0], cf[1]);
        q0.y = pk_bf16(cf[2], cf[3]);
        q0.z = pk_bf16(cf[4], cf[5]);
        q0.w = pk_bf16(cf[6], cf[7]);
        q1.x = pk_bf16(sf[0], sf[1]);
        q1.y = pk_bf16(sf[2], sf[3]);
        q1.z = pk_bf16(sf[4], sf[5]);
        q1.w = pk_bf16(sf[6], sf[7]);
        uint4* dst = (uint4*)&As[b][fr * A_STRIDE + fc * 16];
        dst[0] = q0;    // K_local fc*16 + 0..7  (cos k=1..8)
        dst[1] = q1;    // K_local fc*16 + 8..15 (sin k=1..8)
    };

    // Prologue: features for step 0; start the pipelined x load for step 1.
    featBody(xp[0], 0);
    float xNext = (nsteps > 1) ? xp[2] : 0.0f;   // x for step 1 (channel stride 2)

    const int halfbase = (lane >= 16) ? 8 : 0;   // ISA 16-bit A/B lane->K map

    for (int s = 0; s < nsteps; ++s) {
        __syncthreads();
        if (s + 1 < nsteps) {
            float xc = xNext;
            if (s + 2 < nsteps) xNext = xp[(s + 2) * 2];   // issue load for s+2 now
            featBody(xc, (s + 1) & 1);
        }

        // --- B fragments: contiguous 32B per lane, fully coalesced ---
        BFragU bfr[2];
#pragma unroll
        for (int nt = 0; nt < 2; ++nt) {
            long nb = (col0 >> 4) + nwave * 2 + nt;
            const uint4* p =
                (const uint4*)(W + (((long)s * nbCount) + nb) * 512) + lane * 2;
            bfr[nt].q[0] = p[0];
            bfr[nt].q[1] = p[1];
        }
        if (s + 1 < nsteps) {   // pull next step's weight fragments toward L2/L0
            long nb = (col0 >> 4) + nwave * 2;
            __builtin_prefetch(
                W + (((long)(s + 1) * nbCount) + nb) * 512 + lane * 16, 0, 1);
        }

        // --- A fragments from LDS (two b128 per lane, bank-conflict free) ---
        const int buf = s & 1;
        BFragU afr[4];
#pragma unroll
        for (int mt = 0; mt < 4; ++mt) {
            int row = mwave * 64 + mt * 16 + (lane & 15);
            const unsigned short* base = &As[buf][row * A_STRIDE + halfbase];
            afr[mt].q[0] = *(const uint4*)(base);        // K halfbase..+7
            afr[mt].q[1] = *(const uint4*)(base + 16);   // K halfbase+16..+23
        }

        // --- 8 WMMA per wave per K-step ---
#pragma unroll
        for (int mt = 0; mt < 4; ++mt)
#pragma unroll
            for (int nt = 0; nt < 2; ++nt)
                acc[mt][nt] = __builtin_amdgcn_wmma_f32_16x16x32_bf16(
                    false, afr[mt].v, false, bfr[nt].v,
                    (short)0, acc[mt][nt], false, false);
    }

    // --- Epilogue: bias add + store (each VGPR row -> two 64B segments) ---
#pragma unroll
    for (int nt = 0; nt < 2; ++nt) {
        int col = col0 + nwave * 32 + nt * 16 + (lane & 15);
        float bv = bias[col];
#pragma unroll
        for (int mt = 0; mt < 4; ++mt) {
            int rbase = row0 + mwave * 64 + mt * 16 + ((lane >= 16) ? 8 : 0);
#pragma unroll
            for (int r = 0; r < 8; ++r)
                Y[(long)(rbase + r) * N + col] = acc[mt][nt][r] + bv;
        }
    }
}

// ---------------------------------------------------------------------------
extern "C" void kernel_launch(void* const* d_in, const int* in_sizes, int n_in,
                              void* d_out, int out_size, void* d_ws, size_t ws_size,
                              hipStream_t stream) {
    (void)in_sizes; (void)n_in; (void)out_size; (void)ws_size;
    const float* x  = (const float*)d_in[0];
    const float* C1 = (const float*)d_in[1];
    const float* b1 = (const float*)d_in[2];
    const float* C2 = (const float*)d_in[3];
    const float* b2 = (const float*)d_in[4];
    const float* C3 = (const float*)d_in[5];
    const float* b3 = (const float*)d_in[6];
    float* out = (float*)d_out;

    // Workspace layout (all 16B aligned): W1|W2|W3 (bf16) then Y1|Y2 (f32)
    unsigned short* W1 = (unsigned short*)d_ws;          // 12288 x 2048
    size_t w1 = (size_t)12288 * 2048;
    unsigned short* W2 = W1 + w1;                        // 32768 x 2048
    size_t w2 = (size_t)32768 * 2048;
    unsigned short* W3 = W2 + w2;                        // 32768 x 512
    size_t w3 = (size_t)32768 * 512;
    float* Y1 = (float*)(W3 + w3);                       // 4096 x 2048
    float* Y2 = Y1 + (size_t)KAN_BATCH * 2048;           // 4096 x 2048

    auto conv = [&](const float* C, unsigned short* W, int IN, int OUT) {
        long total = (long)IN * OUT;
        int blocks = (int)((total + 255) / 256);
        convert_weights<<<blocks, 256, 0, stream>>>(C, W, IN, OUT);
    };
    conv(C1, W1, 768, 2048);
    conv(C2, W2, 2048, 2048);
    conv(C3, W3, 2048, 512);

    kan_layer_wmma<<<dim3(2048 / BN, KAN_BATCH / BM), 256, 0, stream>>>(
        x, W1, b1, Y1, 768, 2048);
    kan_layer_wmma<<<dim3(2048 / BN, KAN_BATCH / BM), 256, 0, stream>>>(
        Y1, W2, b2, Y2, 2048, 2048);
    kan_layer_wmma<<<dim3(512 / BN, KAN_BATCH / BM), 256, 0, stream>>>(
        Y2, W3, b3, out, 2048, 512);
}
